// TrafficModel_71390946394304
// MI455X (gfx1250) — matrix operand order repaired and checked by reading, not compiled, not run on hardware
//
#include <hip/hip_runtime.h>
#include <hip/hip_bf16.h>

typedef __attribute__((ext_vector_type(16))) __bf16 v16bf;
typedef __attribute__((ext_vector_type(8)))  float  v8f;

#define R_DIM 128
#define T_DIM 4096
#define M_DIM 32768
#define N_DIM 4096
#define L_DIM 16384
#define ALPHA_LO 0.01f
#define ALPHA_HI 1.0f
#define BETA_LO  1.0f
#define BETA_HI  8.0f

#define CHUNK 512
#define NCH   (L_DIM / CHUNK)   // 32 L-chunks for the partial reduction

// ---------------------------------------------------------------- zero fill
__global__ void k_zero(float4* __restrict__ p, long n) {
    long i  = (long)blockIdx.x * blockDim.x + threadIdx.x;
    long st = (long)gridDim.x * blockDim.x;
    float4 z = make_float4(0.f, 0.f, 0.f, 0.f);
    for (; i < n; i += st) p[i] = z;
}

// ------------------------------------------------- scalar + per-row prep
// base = sum(t0*radio); alpha/beta from raw logits; w = t0*alpha*radio; 1/cap
__global__ void k_prep(const float* __restrict__ t0, const float* __restrict__ cap,
                       const float* __restrict__ radio,
                       const float* __restrict__ alpha_raw, const float* __restrict__ beta_raw,
                       float* __restrict__ w, float* __restrict__ inv_cap,
                       float* __restrict__ scal) {
    __shared__ float red[256];
    int tid = threadIdx.x;
    float asig = 1.f / (1.f + __expf(-alpha_raw[0]));
    float bsig = 1.f / (1.f + __expf(-beta_raw[0]));
    float alpha = ALPHA_LO + (ALPHA_HI - ALPHA_LO) * asig;
    float beta  = BETA_LO  + (BETA_HI  - BETA_LO)  * bsig;
    float acc = 0.f;
    for (int l = tid; l < L_DIM; l += 256) {
        float t0l = t0[l], rl = radio[l];
        acc += t0l * rl;
        w[l] = t0l * alpha * rl;
        inv_cap[l] = 1.0f / cap[l];
    }
    red[tid] = acc;
    __syncthreads();
    for (int s = 128; s > 0; s >>= 1) {
        if (tid < s) red[tid] += red[tid + s];
        __syncthreads();
    }
    if (tid == 0) { scal[0] = red[0]; scal[1] = alpha; scal[2] = beta; }
}

// ---------------------------------------------------------------- relu(x)
__global__ void k_relu(const float* __restrict__ x, float* __restrict__ x_pos, int n) {
    int i = blockIdx.x * blockDim.x + threadIdx.x;
    if (i < n) x_pos[i] = fmaxf(x[i], 0.f);
}

// ------------------------------------------------ WMMA GEMM + segment-sum
// One wave32 computes a 16(M) x 64(T) strip of c_pred = A @ relu(x) as four
// 16x16 WMMA tiles sharing one A fragment (bf16x3 split products on
// v_wmma_f32_16x16x32_bf16 -> near-f32 accuracy). The 8 waves of a block
// cover 8 consecutive M-tiles over the SAME 64 T-columns, so B loads hit
// the same L0 lines. Epilogue stores c_pred and atomically scatters into
// the L2-resident node_flow table (fused segment_sum).
__global__ void __launch_bounds__(256) k_gemm(const float* __restrict__ A,
                                              const float* __restrict__ x,
                                              const int*   __restrict__ node_ids,
                                              float* __restrict__ c_pred,
                                              float* __restrict__ nf) {
    int waveId = threadIdx.x >> 5;
    int lane   = threadIdx.x & 31;
    int bt = blockIdx.x & 63;            // 64 T-supertiles (64 columns each)
    int bm = blockIdx.x >> 6;            // 256 M-superblocks (8 tiles = 128 rows)
    int tm  = (bm * 8 + waveId) * 16;
    int tt0 = bt * 64;
    int mr = lane & 15;                  // A-row within tile / B-column within tile
    int h  = lane >> 4;                  // lane half selects K sub-pattern

    const float* Arow = A + (size_t)(tm + mr) * R_DIM;

    v8f acc[4] = {v8f{}, v8f{}, v8f{}, v8f{}};

    for (int kb = 0; kb < R_DIM; kb += 32) {
        if (kb + 32 < R_DIM)
            __builtin_prefetch(Arow + kb + 32, 0, 0);   // global_prefetch next A chunk

        // --- A fragment (16x32 bf16 layout): j<8 -> k=kb+8h+j ; j>=8 -> k=kb+16+8h+(j-8)
        float af[16];
        const float* ap0 = Arow + kb + 8 * h;
        const float* ap1 = Arow + kb + 16 + 8 * h;
        float4 q0 = *(const float4*)(ap0);
        float4 q1 = *(const float4*)(ap0 + 4);
        float4 q2 = *(const float4*)(ap1);
        float4 q3 = *(const float4*)(ap1 + 4);
        af[0]=q0.x;  af[1]=q0.y;  af[2]=q0.z;  af[3]=q0.w;
        af[4]=q1.x;  af[5]=q1.y;  af[6]=q1.z;  af[7]=q1.w;
        af[8]=q2.x;  af[9]=q2.y;  af[10]=q2.z; af[11]=q2.w;
        af[12]=q3.x; af[13]=q3.y; af[14]=q3.z; af[15]=q3.w;

        v16bf ah, al;
        #pragma unroll
        for (int j = 0; j < 16; ++j) {
            __bf16 hi = (__bf16)af[j];
            ah[j] = hi;
            al[j] = (__bf16)(af[j] - (float)hi);
        }

        // --- 4 B fragments (32x16 bf16 layout): j<8 -> k=kb+8h+j ; j>=8 -> k=kb+16+8h+(j-8)
        #pragma unroll
        for (int c = 0; c < 4; ++c) {
            const float* Xcol = x + tt0 + c * 16 + mr;
            float bfv[16];
            #pragma unroll
            for (int j = 0; j < 16; ++j) {
                float v = Xcol[(size_t)(kb + 8 * h + j + (j >= 8 ? 8 : 0)) * T_DIM];
                bfv[j] = fmaxf(v, 0.f);                 // fused relu(x)
            }
            v16bf bh, bl;
            #pragma unroll
            for (int j = 0; j < 16; ++j) {
                __bf16 hi = (__bf16)bfv[j];
                bh[j] = hi;
                bl[j] = (__bf16)(bfv[j] - (float)hi);
            }
            acc[c] = __builtin_amdgcn_wmma_f32_16x16x32_bf16(false, ah, false, bh, (short)0, acc[c], false, false);
            acc[c] = __builtin_amdgcn_wmma_f32_16x16x32_bf16(false, ah, false, bl, (short)0, acc[c], false, false);
            acc[c] = __builtin_amdgcn_wmma_f32_16x16x32_bf16(false, al, false, bh, (short)0, acc[c], false, false);
        }
    }

    // Epilogue: C layout -> acc[c][v] = C[m = v + 8h][n = mr] of tile (tm, tt0+16c)
    #pragma unroll
    for (int v = 0; v < 8; ++v) {
        int m = tm + v + 8 * h;
        int node = node_ids[m];
        float* crow = c_pred + (size_t)m * T_DIM;
        float* nrow = nf + (size_t)node * T_DIM;
        #pragma unroll
        for (int c = 0; c < 4; ++c) {
            int t = tt0 + c * 16 + mr;
            float val = acc[c][v];
            crow[t] = val;
            atomicAdd(&nrow[t], val);     // fused segment_sum, L2-resident
        }
    }
}

// -------------------------------------------- pow + partial reduce over L
// Block (bx, c): threads own one t each; chunk c covers CHUNK rows of L with
// metadata staged in LDS. Gathers hit the 64MB L2-resident node_flow table.
__global__ void __launch_bounds__(256) k_pow(const float* __restrict__ nf,
                                             const int*   __restrict__ row_node,
                                             const float* __restrict__ w,
                                             const float* __restrict__ inv_cap,
                                             const float* __restrict__ scal,
                                             float* __restrict__ partial) {
    __shared__ int   s_node[CHUNK];
    __shared__ float s_w[CHUNK];
    __shared__ float s_ic[CHUNK];
    int tid = threadIdx.x;
    int t = blockIdx.x * 256 + tid;
    int c = blockIdx.y;
    int l0 = c * CHUNK;
    for (int i = tid; i < CHUNK; i += 256) {
        s_node[i] = row_node[l0 + i];
        s_w[i]    = w[l0 + i];
        s_ic[i]   = inv_cap[l0 + i];
    }
    __syncthreads();
    float beta = scal[2];
    float acc = 0.f;
    for (int i = 0; i < CHUNK; ++i) {
        float flow  = nf[(size_t)s_node[i] * T_DIM + t];
        float ratio = fmaxf(flow * s_ic[i], 1e-6f);
        acc += s_w[i] * __expf(beta * __logf(ratio));   // ratio^beta (v_log/v_exp)
    }
    partial[(size_t)c * T_DIM + t] = acc;
}

// --------------------------------------- deterministic final reduce into y
__global__ void k_red(const float* __restrict__ partial, const float* __restrict__ scal,
                      float* __restrict__ y) {
    int t = blockIdx.x * blockDim.x + threadIdx.x;
    if (t >= T_DIM) return;
    float acc = scal[0];                      // base = sum(t0*radio)
    for (int c = 0; c < NCH; ++c) acc += partial[(size_t)c * T_DIM + t];
    y[t] = acc;
}

extern "C" void kernel_launch(void* const* d_in, const int* in_sizes, int n_in,
                              void* d_out, int out_size, void* d_ws, size_t ws_size,
                              hipStream_t stream) {
    const float* x         = (const float*)d_in[0];
    const float* A         = (const float*)d_in[1];
    const float* t0        = (const float*)d_in[2];
    const float* cap       = (const float*)d_in[3];
    const float* radio     = (const float*)d_in[4];
    const float* alpha_raw = (const float*)d_in[5];
    const float* beta_raw  = (const float*)d_in[6];
    const int*   node_ids  = (const int*)d_in[7];
    const int*   row_node  = (const int*)d_in[8];

    float* out    = (float*)d_out;
    float* x_pos  = out;                                    // [R*T]
    float* c_pred = out + (size_t)R_DIM * T_DIM;            // [M*T]
    float* y      = c_pred + (size_t)M_DIM * T_DIM;         // [T]

    float* ws      = (float*)d_ws;
    float* nf      = ws;                                    // [N*T] node_flow (64 MB)
    float* w       = ws + (size_t)N_DIM * T_DIM;            // [L]
    float* inv_cap = w + L_DIM;                             // [L]
    float* partial = inv_cap + L_DIM;                       // [NCH*T]
    float* scal    = partial + (size_t)NCH * T_DIM;         // [3] base, alpha, beta

    // 1) zero node_flow
    k_zero<<<4096, 256, 0, stream>>>((float4*)nf, ((long)N_DIM * T_DIM) / 4);
    // 2) scalars + per-L metadata
    k_prep<<<1, 256, 0, stream>>>(t0, cap, radio, alpha_raw, beta_raw, w, inv_cap, scal);
    // 3) x_pos output
    k_relu<<<(R_DIM * T_DIM) / 256, 256, 0, stream>>>(x, x_pos, R_DIM * T_DIM);
    // 4) WMMA GEMM (16x64 strip per wave) + fused segment-sum scatter
    //    grid = (M/128 M-superblocks) x (T/64 T-supertiles) = 256*64 blocks
    k_gemm<<<256 * 64, 256, 0, stream>>>(A, x, node_ids, c_pred, nf);
    // 5) BPR power term, partial-reduced over L in NCH chunks
    dim3 gp(T_DIM / 256, NCH);
    k_pow<<<gp, 256, 0, stream>>>(nf, row_node, w, inv_cap, scal, partial);
    // 6) deterministic final reduction into y
    k_red<<<T_DIM / 256, 256, 0, stream>>>(partial, scal, y);
}
